// Generator_3066606649895
// MI455X (gfx1250) — matrix operand order repaired and checked by reading, not compiled, not run on hardware
//
#include <hip/hip_runtime.h>
#include <hip/hip_bf16.h>

// Problem constants (match reference).
constexpr int B_ = 16384;
constexpr int H_ = 1024;
constexpr int O_ = 512;

// LDS A-tile row stride in bf16 elements (32 data + 8 pad -> 80B rows,
// fragment b128 reads hit bank bases spaced by 4 -> conflict-free).
constexpr int SROW = 40;

typedef __attribute__((ext_vector_type(16))) __bf16 v16bf;
typedef __attribute__((ext_vector_type(8)))  __bf16 v8bf;
typedef __attribute__((ext_vector_type(8)))  float  v8f;

// Workspace layout (bf16 elements).
constexpr size_t N_WIH  = (size_t)3 * H_ * O_;
constexpr size_t N_WHH  = (size_t)3 * H_ * H_;
constexpr size_t N_WOUT = (size_t)O_ * H_;
constexpr size_t N_INP  = (size_t)B_ * O_;
constexpr size_t N_HID  = (size_t)B_ * H_;

// ---------------------------------------------------------------------------
// f32 -> bf16 bulk converter: 8 elements per thread, b128 in / b128 out.
// ---------------------------------------------------------------------------
__global__ __launch_bounds__(256) void cvt_f32_bf16(
    const float* __restrict__ src, __bf16* __restrict__ dst, long n8)
{
    long i = (long)blockIdx.x * blockDim.x + threadIdx.x;
    if (i >= n8) return;
    const float4* s = reinterpret_cast<const float4*>(src) + 2 * i;
    float4 a = s[0], b = s[1];
    v8bf d;
    d[0] = (__bf16)a.x; d[1] = (__bf16)a.y; d[2] = (__bf16)a.z; d[3] = (__bf16)a.w;
    d[4] = (__bf16)b.x; d[5] = (__bf16)b.y; d[6] = (__bf16)b.z; d[7] = (__bf16)b.w;
    reinterpret_cast<v8bf*>(dst)[i] = d;
}

// ---------------- CDNA5 async Global->LDS helpers (ISA 08_async_tensor §4) ----

__device__ __forceinline__ void async_load_b128(const __bf16* gaddr, unsigned lds_addr) {
    // VDST = LDS byte address VGPR, VADDR = 64-bit global address.
    asm volatile("global_load_async_to_lds_b128 %0, %1, off"
                 :: "v"(lds_addr), "v"(gaddr) : "memory");
}
__device__ __forceinline__ void wait_async_le1() {
    asm volatile("s_wait_asynccnt 0x1" ::: "memory");
}
__device__ __forceinline__ void wait_async_le0() {
    asm volatile("s_wait_asynccnt 0x0" ::: "memory");
}

// Stage one 32(rows) x 32(K) bf16 A-chunk into LDS: 128 threads x one b128.
__device__ __forceinline__ void stage_tile(const __bf16* __restrict__ blk, size_t stride,
                                           int k0, __bf16* sbuf) {
    const int t   = threadIdx.x;   // 0..127, uniform control flow
    const int row = t >> 2;
    const int seg = t & 3;
    const __bf16* g = blk + (size_t)row * stride + k0 + seg * 8;
    unsigned l = (unsigned)(uintptr_t)(sbuf + row * SROW + seg * 8);
    async_load_b128(g, l);
}

// ---- bf16 operand loaders (layouts per CDNA5 ISA §7.12.2) ----

// A fragment from the staged LDS tile: lane (m = lane&15, g = lane>>4),
// M-subtile `sub`: elements 0..7 -> Kloc 8g..8g+7, 8..15 -> Kloc 16+8g..+7.
__device__ __forceinline__ v16bf load_a_lds(const __bf16* sbuf, int sub, int m16, int g) {
    const __bf16* r = sbuf + (m16 + 16 * sub) * SROW;
    v8bf lo = *reinterpret_cast<const v8bf*>(r + 8 * g);
    v8bf hi = *reinterpret_cast<const v8bf*>(r + 16 + 8 * g);
    return __builtin_shufflevector(lo, hi, 0, 1, 2, 3, 4, 5, 6, 7,
                                           8, 9, 10, 11, 12, 13, 14, 15);
}

// A directly from global (kernel 2).
__device__ __forceinline__ v16bf load_a16(const __bf16* __restrict__ row, int k0, int g) {
    v8bf lo = *reinterpret_cast<const v8bf*>(row + k0 + 8 * g);
    v8bf hi = *reinterpret_cast<const v8bf*>(row + k0 + 16 + 8 * g);
    return __builtin_shufflevector(lo, hi, 0, 1, 2, 3, 4, 5, 6, 7,
                                           8, 9, 10, 11, 12, 13, 14, 15);
}

// B-matrix 32x16 bf16, lane (n = lane&15, g = lane>>4): elements 0..7 -> K = k0 + 8g..8g+7, 8..15 -> K = k0 + 16+8g..+7
__device__ __forceinline__ v16bf load_b16(const __bf16* __restrict__ row, int k0, int g) {
    v8bf lo = *reinterpret_cast<const v8bf*>(row + k0 + 8 * g);
    v8bf hi = *reinterpret_cast<const v8bf*>(row + k0 + 16 + 8 * g);
    return __builtin_shufflevector(lo, hi, 0, 1, 2, 3, 4, 5, 6, 7,
                                           8, 9, 10, 11, 12, 13, 14, 15);
}

__device__ __forceinline__ v8f wmma_bf16(v16bf a, v16bf b, v8f c) {
    return __builtin_amdgcn_wmma_f32_16x16x32_bf16(false, a, false, b, (short)0, c, false, false);
}

__device__ __forceinline__ float sigmoidf_(float x) {
    return 1.0f / (1.0f + __expf(-x));
}

// ---------------------------------------------------------------------------
// Kernel 1: fused GRU cell, bf16 operands, f32 accumulate.
// Block = 128 threads (4 waves) covering a 32(M) x 64(N) region; the shared
// 32x32 A-chunk is staged once per block into LDS via async Global->LDS DMA
// (double-buffered); each wave owns one 16-column tile with 8 accumulators
// (r/z shared across the input-path K=O and hidden-path K=H GEMMs).
// grid = (H/64, B/32).
// ---------------------------------------------------------------------------
__global__ __launch_bounds__(128) void gru_cell_wmma(
    const __bf16* __restrict__ inp_bf,  // [B, O]
    const __bf16* __restrict__ hid_bf,  // [B, H]
    const float*  __restrict__ hidden,  // [B, H] f32 (for the z*h blend)
    const __bf16* __restrict__ w_ih,    // [3H, O] bf16
    const __bf16* __restrict__ w_hh,    // [3H, H] bf16
    const float*  __restrict__ b_ih,    // [3H]
    const float*  __restrict__ b_hh,    // [3H]
    float*  __restrict__ h_new,         // [B, H] f32 (tuple output)
    __bf16* __restrict__ hnew_bf)       // [B, H] bf16 (operand for kernel 2)
{
    __shared__ __align__(16) __bf16 sA[2][32 * SROW];

    const int lane = threadIdx.x & 31;
    const int wave = threadIdx.x >> 5;
    const int g    = lane >> 4;
    const int m16  = lane & 15;

    const int rb = blockIdx.y * 32;             // batch-row tile base (2 subtiles)
    const int cb = blockIdx.x * 64 + wave * 16; // hidden-col tile base
    const int j  = cb + m16;                    // this lane's output column

    const __bf16* Ablk0 = inp_bf + (size_t)rb * O_;   // stride O_
    const __bf16* Ablk1 = hid_bf + (size_t)rb * H_;   // stride H_

    const __bf16* wih_r = w_ih + (size_t)(0 * H_ + j) * O_;
    const __bf16* wih_z = w_ih + (size_t)(1 * H_ + j) * O_;
    const __bf16* wih_n = w_ih + (size_t)(2 * H_ + j) * O_;
    const __bf16* whh_r = w_hh + (size_t)(0 * H_ + j) * H_;
    const __bf16* whh_z = w_hh + (size_t)(1 * H_ + j) * H_;
    const __bf16* whh_n = w_hh + (size_t)(2 * H_ + j) * H_;

    v8f ar0 = {}, ar1 = {};   // i_r + h_r
    v8f az0 = {}, az1 = {};   // i_z + h_z
    v8f ai0 = {}, ai1 = {};   // i_n
    v8f ah0 = {}, ah1 = {};   // h_n

    int buf = 0;
    stage_tile(Ablk0, O_, 0, sA[0]);

    // Input path: K = O = 512.
    for (int k0 = 0; k0 < O_; k0 += 32) {
        if (k0 + 32 < O_) stage_tile(Ablk0, O_, k0 + 32, sA[buf ^ 1]);
        else              stage_tile(Ablk1, H_, 0,       sA[buf ^ 1]);  // pre-stage phase 2
        wait_async_le1();
        __syncthreads();

        __builtin_prefetch(wih_r + k0 + 128, 0, 1);
        __builtin_prefetch(wih_z + k0 + 128, 0, 1);
        __builtin_prefetch(wih_n + k0 + 128, 0, 1);
        v16bf a0 = load_a_lds(sA[buf], 0, m16, g);
        v16bf a1 = load_a_lds(sA[buf], 1, m16, g);
        v16bf br = load_b16(wih_r, k0, g);
        v16bf bz = load_b16(wih_z, k0, g);
        v16bf bn = load_b16(wih_n, k0, g);
        ar0 = wmma_bf16(a0, br, ar0);  ar1 = wmma_bf16(a1, br, ar1);
        az0 = wmma_bf16(a0, bz, az0);  az1 = wmma_bf16(a1, bz, az1);
        ai0 = wmma_bf16(a0, bn, ai0);  ai1 = wmma_bf16(a1, bn, ai1);

        __syncthreads();   // everyone done with sA[buf] before it is restaged
        buf ^= 1;
    }

    // Hidden path: K = H = 1024 (chunk 0 already staged into sA[buf]).
    for (int k0 = 0; k0 < H_; k0 += 32) {
        if (k0 + 32 < H_) { stage_tile(Ablk1, H_, k0 + 32, sA[buf ^ 1]); wait_async_le1(); }
        else              { wait_async_le0(); }
        __syncthreads();

        __builtin_prefetch(whh_r + k0 + 128, 0, 1);
        __builtin_prefetch(whh_z + k0 + 128, 0, 1);
        __builtin_prefetch(whh_n + k0 + 128, 0, 1);
        v16bf a0 = load_a_lds(sA[buf], 0, m16, g);
        v16bf a1 = load_a_lds(sA[buf], 1, m16, g);
        v16bf br = load_b16(whh_r, k0, g);
        v16bf bz = load_b16(whh_z, k0, g);
        v16bf bn = load_b16(whh_n, k0, g);
        ar0 = wmma_bf16(a0, br, ar0);  ar1 = wmma_bf16(a1, br, ar1);
        az0 = wmma_bf16(a0, bz, az0);  az1 = wmma_bf16(a1, bz, az1);
        ah0 = wmma_bf16(a0, bn, ah0);  ah1 = wmma_bf16(a1, bn, ah1);

        __syncthreads();
        buf ^= 1;
    }

    // Gate math. C/D layout: element v <-> (row base + v + 8*g, col j).
    const float brz  = b_ih[0 * H_ + j] + b_hh[0 * H_ + j];
    const float bzz  = b_ih[1 * H_ + j] + b_hh[1 * H_ + j];
    const float bihn = b_ih[2 * H_ + j];
    const float bhhn = b_hh[2 * H_ + j];

#pragma unroll
    for (int t = 0; t < 2; ++t) {
        const v8f& accr = t ? ar1 : ar0;
        const v8f& accz = t ? az1 : az0;
        const v8f& acci = t ? ai1 : ai0;
        const v8f& acch = t ? ah1 : ah0;
#pragma unroll
        for (int v = 0; v < 8; ++v) {
            const int brow = rb + t * 16 + v + 8 * g;
            const float r  = sigmoidf_(accr[v] + brz);
            const float z  = sigmoidf_(accz[v] + bzz);
            const float n  = tanhf(acci[v] + bihn + r * (acch[v] + bhhn));
            const float hp = hidden[(size_t)brow * H_ + j];
            const float hn = (1.0f - z) * n + z * hp;
            h_new[(size_t)brow * H_ + j]   = hn;
            hnew_bf[(size_t)brow * H_ + j] = (__bf16)hn;
        }
    }
}

// ---------------------------------------------------------------------------
// Kernel 2: out = tanh(h_new @ w_out.T + b_out). 32(M) x 16(N) per wave.
// grid = (O/64, B/32), block = 128.
// ---------------------------------------------------------------------------
__global__ __launch_bounds__(128) void out_proj_wmma(
    const __bf16* __restrict__ hnew_bf, // [B, H] bf16
    const __bf16* __restrict__ w_out,   // [O, H] bf16
    const float*  __restrict__ b_out,   // [O]
    float* __restrict__ out)            // [B, O]
{
    const int lane = threadIdx.x & 31;
    const int wave = threadIdx.x >> 5;
    const int g    = lane >> 4;
    const int m16  = lane & 15;

    const int rb = blockIdx.y * 32;
    const int cb = blockIdx.x * 64 + wave * 16;
    const int j  = cb + m16;

    const __bf16* arow0 = hnew_bf + (size_t)(rb + m16) * H_;
    const __bf16* arow1 = arow0 + (size_t)16 * H_;
    const __bf16* wrow  = w_out + (size_t)j * H_;

    v8f acc0 = {}, acc1 = {};
    for (int k0 = 0; k0 < H_; k0 += 32) {
        __builtin_prefetch(wrow + k0 + 128, 0, 1);
        v16bf a0 = load_a16(arow0, k0, g);
        v16bf a1 = load_a16(arow1, k0, g);
        v16bf b  = load_b16(wrow, k0, g);
        acc0 = wmma_bf16(a0, b, acc0);
        acc1 = wmma_bf16(a1, b, acc1);
    }

    const float bo = b_out[j];
#pragma unroll
    for (int t = 0; t < 2; ++t) {
        const v8f& acc = t ? acc1 : acc0;
#pragma unroll
        for (int v = 0; v < 8; ++v) {
            const int brow = rb + t * 16 + v + 8 * g;
            out[(size_t)brow * O_ + j] = tanhf(acc[v] + bo);
        }
    }
}

extern "C" void kernel_launch(void* const* d_in, const int* in_sizes, int n_in,
                              void* d_out, int out_size, void* d_ws, size_t ws_size,
                              hipStream_t stream) {
    (void)in_sizes; (void)n_in; (void)out_size; (void)ws_size;

    const float* inp    = (const float*)d_in[0];  // [B, O]
    const float* hidden = (const float*)d_in[1];  // [1, B, H] -> [B, H]
    const float* w_ih   = (const float*)d_in[2];  // [3H, O]
    const float* w_hh   = (const float*)d_in[3];  // [3H, H]
    const float* b_ih   = (const float*)d_in[4];  // [3H]
    const float* b_hh   = (const float*)d_in[5];  // [3H]
    const float* w_out  = (const float*)d_in[6];  // [O, H]
    const float* b_out  = (const float*)d_in[7];  // [O]

    float* out   = (float*)d_out;                 // [B, O]
    float* h_new = out + (size_t)B_ * O_;         // [B, H]

    __bf16* ws      = (__bf16*)d_ws;
    __bf16* wih_bf  = ws;
    __bf16* whh_bf  = wih_bf  + N_WIH;
    __bf16* wout_bf = whh_bf  + N_WHH;
    __bf16* inp_bf  = wout_bf + N_WOUT;
    __bf16* hid_bf  = inp_bf  + N_INP;
    __bf16* hnew_bf = hid_bf  + N_HID;

    // Stage 0: one-time f32 -> bf16 conversion of all GEMM operands.
    auto cvt = [&](const float* s, __bf16* d, size_t n) {
        long n8 = (long)(n / 8);
        cvt_f32_bf16<<<dim3((unsigned)((n8 + 255) / 256)), 256, 0, stream>>>(s, d, n8);
    };
    cvt(w_ih,   wih_bf,  N_WIH);
    cvt(w_hh,   whh_bf,  N_WHH);
    cvt(w_out,  wout_bf, N_WOUT);
    cvt(inp,    inp_bf,  N_INP);
    cvt(hidden, hid_bf,  N_HID);

    dim3 block(128);
    dim3 grid1(H_ / 64, B_ / 32);  // (16, 512)
    dim3 grid2(O_ / 64, B_ / 32);  // (8, 512)

    gru_cell_wmma<<<grid1, block, 0, stream>>>(inp_bf, hid_bf, hidden, wih_bf, whh_bf,
                                               b_ih, b_hh, h_new, hnew_bf);
    out_proj_wmma<<<grid2, block, 0, stream>>>(hnew_bf, wout_bf, b_out, out);
}
